// DlrmSmall_48576080117969
// MI455X (gfx1250) — matrix operand order repaired and checked by reading, not compiled, not run on hardware
//
#include <hip/hip_runtime.h>

// ---------------------------------------------------------------------------
// DLRM-small forward for gfx1250 (MI455X): bf16 WMMA everywhere GEMM-shaped,
// f32 accumulate, wave32. Matrix math via v_wmma_f32_16x16x32_bf16.
// Fragment loads are vectorized (v8bf -> ds_load_b128); all weight panels are
// staged in LDS transposed so B-fragments are contiguous per lane.
// ---------------------------------------------------------------------------

typedef __attribute__((ext_vector_type(16))) __bf16 v16bf;
typedef __attribute__((ext_vector_type(8)))  __bf16 v8bf;
typedef __attribute__((ext_vector_type(4)))  __bf16 v4bf;
typedef __attribute__((ext_vector_type(8)))  float  v8f;

#define DLRM_B     16384
#define DLRM_VOCAB 1048576
#define DLRM_EMBED 128
#define NSPARSE    26
#define NCOMB      27          // 1 (bottom h) + 26 embeddings
#define FEATPAD    512         // 128 (h) + 378 (triu) + 6 zero pad

__device__ __forceinline__ v8f wmma_bf16(v16bf a, v16bf b, v8f c) {
    // (neg_a, A, neg_b, B, c_mod, C, reuse_a, reuse_b)
    return __builtin_amdgcn_wmma_f32_16x16x32_bf16(false, a, false, b,
                                                   (short)0, c, false, false);
}

__device__ __forceinline__ v16bf cat8(v8bf lo, v8bf hi) {
    return __builtin_shufflevector(lo, hi, 0, 1, 2, 3, 4, 5, 6, 7,
                                           8, 9, 10, 11, 12, 13, 14, 15);
}

// A fragment (16x32, MxK) from bf16 matrix, leading dim lda (ISA 7.12.2):
// lanes 0..15 : M=lane,    elems 0..7 -> K 0..7,  elems 8..15 -> K 16..23
// lanes 16..31: M=lane-16, elems 0..7 -> K 8..15, elems 8..15 -> K 24..31
// Two contiguous 16 B chunks per lane -> 2x ds_load_b128.
__device__ __forceinline__ v16bf load_a_frag(const __bf16* A, int lda,
                                             int row0, int k0, int lane) {
    int r  = row0 + (lane & 15);
    int kb = (lane >> 4) * 8;
    const __bf16* p = A + (size_t)r * lda + k0 + kb;
    v8bf lo = *reinterpret_cast<const v8bf*>(p);
    v8bf hi = *reinterpret_cast<const v8bf*>(p + 16);
    return cat8(lo, hi);
}

// B fragment (32x16, KxN) built from a K-transposed panel T where
// T[n][k] = B[k][n] (leading dim ldt, in elements). Lane layout:
// lanes 0..15 hold K 0..7,16..23; lanes 16..31 hold K 8..15,24..31, N = lane&15.
// Two contiguous 16 B chunks per lane -> 2x ds_load_b128.
__device__ __forceinline__ v16bf load_bt_frag(const __bf16* T, int ldt,
                                              int n0, int k0, int lane) {
    int c  = n0 + (lane & 15);
    int kb = k0 + (lane >> 4) * 8;
    const __bf16* p = T + (size_t)c * ldt + kb;
    v8bf lo = *reinterpret_cast<const v8bf*>(p);
    v8bf hi = *reinterpret_cast<const v8bf*>(p + 16);
    return cat8(lo, hi);
}

// ---------------------------------------------------------------------------
// 1) Bottom MLP: 13 -> 512 -> 256 -> 128, ReLU. 32 rows / block, 2 waves.
//    Weight panels staged transposed in LDS (8 KB), n-chunks of 128.
//    Writes h (bf16) into combined[:,0,:] and feat[:,0:128]; zeros feat 506..511.
//    Static LDS: 2 + 32 + 16 + 8 = 58 KB.
// ---------------------------------------------------------------------------
__global__ __launch_bounds__(64) void bottom_kernel(
    const float* __restrict__ x,
    const float* __restrict__ w0, const float* __restrict__ b0,
    const float* __restrict__ w1, const float* __restrict__ b1,
    const float* __restrict__ w2, const float* __restrict__ b2,
    unsigned short* __restrict__ combined_u16,
    unsigned short* __restrict__ feat_u16) {

    __bf16* combined = reinterpret_cast<__bf16*>(combined_u16);
    __bf16* feat     = reinterpret_cast<__bf16*>(feat_u16);

    __shared__ __bf16 sX[32][32];     //  2 KB dense inputs, K padded 13->32
    __shared__ __bf16 sH1[32][512];   // 32 KB layer-0 activations
    __shared__ __bf16 sH2[32][256];   // 16 KB layer-1 activations
    __shared__ __bf16 sWt[128][32];   //  8 KB transposed weight panel

    const int tid = threadIdx.x, lane = tid & 31, wave = tid >> 5;
    const int rowBase = blockIdx.x * 32;
    const int r0 = wave * 16;
    const int hi = lane >> 4, cl = lane & 15;

    // Stage dense features (zero-padded K).
    for (int idx = tid; idx < 32 * 32; idx += 64) {
        int r = idx >> 5, c = idx & 31;
        float v = (c < 13) ? x[(size_t)(rowBase + r) * 39 + c] : 0.0f;
        sX[r][c] = (__bf16)v;
    }
    __syncthreads();

    // ----- layer 0: K=13 (pad 32), N=512, 4 n-chunks of 128 -----
    {
        v16bf a = load_a_frag(&sX[0][0], 32, r0, 0, lane);
        for (int nc = 0; nc < 4; ++nc) {
            __syncthreads();
            for (int idx = tid; idx < 32 * 128; idx += 64) {
                int k = idx >> 7, n = idx & 127;
                sWt[n][k] = (__bf16)((k < 13)
                    ? w0[(size_t)k * 512 + nc * 128 + n] : 0.0f);
            }
            __syncthreads();
#pragma unroll
            for (int nt = 0; nt < 8; ++nt) {
                v16bf b = load_bt_frag(&sWt[0][0], 32, nt * 16, 0, lane);
                v8f c = {};
                c = wmma_bf16(a, b, c);
                int col = nc * 128 + nt * 16 + cl;
                float bias = b0[col];
#pragma unroll
                for (int q = 0; q < 8; ++q) {
                    float v = fmaxf(c[q] + bias, 0.0f);
                    sH1[r0 + q + hi * 8][col] = (__bf16)v;
                }
            }
        }
    }
    __syncthreads();

    // ----- layer 1: K=512, N=256, 2 n-chunks of 128 -----
    for (int nc = 0; nc < 2; ++nc) {
        v8f acc[8];
        v8f z = {};
#pragma unroll
        for (int i = 0; i < 8; ++i) acc[i] = z;
        for (int ks = 0; ks < 16; ++ks) {
            __syncthreads();
            for (int idx = tid; idx < 32 * 128; idx += 64) {
                int k = idx >> 7, n = idx & 127;
                sWt[n][k] =
                    (__bf16)w1[(size_t)(ks * 32 + k) * 256 + nc * 128 + n];
            }
            __syncthreads();
            v16bf a = load_a_frag(&sH1[0][0], 512, r0, ks * 32, lane);
#pragma unroll
            for (int nt = 0; nt < 8; ++nt) {
                v16bf b = load_bt_frag(&sWt[0][0], 32, nt * 16, 0, lane);
                acc[nt] = wmma_bf16(a, b, acc[nt]);
            }
        }
#pragma unroll
        for (int nt = 0; nt < 8; ++nt) {
            int col = nc * 128 + nt * 16 + cl;
            float bias = b1[col];
#pragma unroll
            for (int q = 0; q < 8; ++q) {
                float v = fmaxf(acc[nt][q] + bias, 0.0f);
                sH2[r0 + q + hi * 8][col] = (__bf16)v;
            }
        }
    }
    __syncthreads();

    // ----- layer 2: K=256, N=128, single n-chunk -----
    {
        v8f acc[8];
        v8f z = {};
#pragma unroll
        for (int i = 0; i < 8; ++i) acc[i] = z;
        for (int ks = 0; ks < 8; ++ks) {
            __syncthreads();
            for (int idx = tid; idx < 32 * 128; idx += 64) {
                int k = idx >> 7, n = idx & 127;
                sWt[n][k] = (__bf16)w2[(size_t)(ks * 32 + k) * 128 + n];
            }
            __syncthreads();
            v16bf a = load_a_frag(&sH2[0][0], 256, r0, ks * 32, lane);
#pragma unroll
            for (int nt = 0; nt < 8; ++nt) {
                v16bf b = load_bt_frag(&sWt[0][0], 32, nt * 16, 0, lane);
                acc[nt] = wmma_bf16(a, b, acc[nt]);
            }
        }
#pragma unroll
        for (int nt = 0; nt < 8; ++nt) {
            int col = nt * 16 + cl;
            float bias = b2[col];
#pragma unroll
            for (int q = 0; q < 8; ++q) {
                int grow = rowBase + r0 + q + hi * 8;
                float v = fmaxf(acc[nt][q] + bias, 0.0f);
                __bf16 h = (__bf16)v;
                feat[(size_t)grow * FEATPAD + col] = h;
                combined[(size_t)grow * NCOMB * DLRM_EMBED + col] = h;  // row 0
            }
        }
    }

    // zero-pad feat cols 506..511 for this block's rows
    for (int idx = tid; idx < 32 * 6; idx += 64) {
        int r = idx / 6, c = 506 + idx % 6;
        feat[(size_t)(rowBase + r) * FEATPAD + c] = (__bf16)0.0f;
    }
}

// ---------------------------------------------------------------------------
// 2) Embedding gather: one wave per (sample, table). 512 B / row, coalesced
//    float4/lane; f32 -> bf16, packed v4bf store (global_store_b64).
// ---------------------------------------------------------------------------
__global__ __launch_bounds__(256) void gather_kernel(
    const float* __restrict__ x, const float* __restrict__ emb,
    unsigned short* __restrict__ combined_u16) {

    __bf16* combined = reinterpret_cast<__bf16*>(combined_u16);
    int wid  = blockIdx.x * 8 + (threadIdx.x >> 5);
    int lane = threadIdx.x & 31;
    int b = wid / NSPARSE, j = wid - b * NSPARSE;

    int idx = ((int)x[(size_t)b * 39 + 13 + j]) & (DLRM_VOCAB - 1);
    const float4 v4 =
        reinterpret_cast<const float4*>(emb + (size_t)idx * DLRM_EMBED)[lane];

    v4bf h;
    h[0] = (__bf16)v4.x;
    h[1] = (__bf16)v4.y;
    h[2] = (__bf16)v4.z;
    h[3] = (__bf16)v4.w;
    __bf16* dst = combined + ((size_t)b * NCOMB + 1 + j) * DLRM_EMBED + lane * 4;
    *reinterpret_cast<v4bf*>(dst) = h;
}

// ---------------------------------------------------------------------------
// 3) Interaction: per sample G = C C^T, C = combined (27x128, padded to 32).
//    4 samples / block (1 wave each), 2x2 WMMA tiles, K=128 in 4 steps.
//    Both A and B fragments come from the same row-major LDS tile (B = C^T).
// ---------------------------------------------------------------------------
__global__ __launch_bounds__(128) void interact_kernel(
    const unsigned short* __restrict__ combined_u16,
    unsigned short* __restrict__ feat_u16) {

    const __bf16* combined = reinterpret_cast<const __bf16*>(combined_u16);
    __bf16* feat           = reinterpret_cast<__bf16*>(feat_u16);

    __shared__ __bf16 sC[4][32][DLRM_EMBED];  // 32 KB, rows 27..31 zeroed

    const int tid = threadIdx.x, lane = tid & 31, wave = tid >> 5;
    const int s0 = blockIdx.x * 4;

    for (int idx = tid; idx < 4 * 32 * DLRM_EMBED; idx += 128) {
        int s = idx >> 12, r = (idx >> 7) & 31, c = idx & 127;
        sC[s][r][c] = (r < NCOMB)
            ? combined[((size_t)(s0 + s) * NCOMB + r) * DLRM_EMBED + c]
            : (__bf16)0.0f;
    }
    __syncthreads();

    const int s = wave;
    v8f acc[4];
    v8f z = {};
#pragma unroll
    for (int i = 0; i < 4; ++i) acc[i] = z;

    for (int ks = 0; ks < 4; ++ks) {
        int k0 = ks * 32;
        v16bf fA0 = load_a_frag(&sC[s][0][0], DLRM_EMBED, 0,  k0, lane);
        v16bf fA1 = load_a_frag(&sC[s][0][0], DLRM_EMBED, 16, k0, lane);
        v16bf fB0 = load_bt_frag(&sC[s][0][0], DLRM_EMBED, 0,  k0, lane);
        v16bf fB1 = load_bt_frag(&sC[s][0][0], DLRM_EMBED, 16, k0, lane);
        acc[0] = wmma_bf16(fA0, fB0, acc[0]);
        acc[1] = wmma_bf16(fA0, fB1, acc[1]);
        acc[2] = wmma_bf16(fA1, fB0, acc[2]);
        acc[3] = wmma_bf16(fA1, fB1, acc[3]);
    }

    __bf16* frow = feat + (size_t)(s0 + s) * FEATPAD + DLRM_EMBED;
    const int hi = lane >> 4, cl = lane & 15;
#pragma unroll
    for (int t = 0; t < 4; ++t) {
        int ti = t >> 1, tj = t & 1;
#pragma unroll
        for (int q = 0; q < 8; ++q) {
            int m = ti * 16 + q + hi * 8;
            int n = tj * 16 + cl;
            if (m < NCOMB && n < NCOMB && n >= m) {
                int flat = m * NCOMB - (m * (m - 1)) / 2 + (n - m);
                frow[flat] = (__bf16)(acc[t][q]);
            }
        }
    }
}

// ---------------------------------------------------------------------------
// 4) Generic top-MLP GEMM: out = relu(A(bf16, MxKpad) @ W(f32, KxN) + bias).
//    Block = 128 rows x 64 cols, 8 waves (16 rows x 4 n-tiles each).
//    A panel (8 KB) and K-transposed W panel (4 KB) staged in LDS per k-step.
//    Next A panel prefetched (global_prefetch_b8) during current k-step.
// ---------------------------------------------------------------------------
__global__ __launch_bounds__(256) void gemm_kernel(
    const unsigned short* __restrict__ A_u16, int lda,
    const float* __restrict__ W, const float* __restrict__ bias,
    int Kreal, int Ksteps, int N,
    unsigned short* __restrict__ out_u16, int ldo, int do_relu) {

    const __bf16* A = reinterpret_cast<const __bf16*>(A_u16);
    __bf16* out     = reinterpret_cast<__bf16*>(out_u16);

    __shared__ __bf16 sA[128][32];   // 8 KB
    __shared__ __bf16 sWt[64][32];   // 4 KB, transposed: sWt[n][k] = W[k][n]

    const int tid = threadIdx.x, lane = tid & 31, wave = tid >> 5;
    const int rowBase = blockIdx.x * 128;
    const int nBase   = blockIdx.y * 64;

    v8f acc[4];
    v8f z = {};
#pragma unroll
    for (int i = 0; i < 4; ++i) acc[i] = z;

    for (int ks = 0; ks < Ksteps; ++ks) {
        int k0 = ks * 32;
        __syncthreads();
        for (int idx = tid; idx < 128 * 32; idx += 256) {
            int r = idx >> 5, c = idx & 31;
            sA[r][c] = A[(size_t)(rowBase + r) * lda + k0 + c];
        }
        for (int idx = tid; idx < 32 * 64; idx += 256) {
            int k = idx >> 6, n = idx & 63;
            int gk = k0 + k;
            sWt[n][k] = (__bf16)((gk < Kreal)
                                     ? W[(size_t)gk * N + nBase + n]
                                     : 0.0f);
        }
        // prefetch next k-step A panel while this one is consumed
        if (ks + 1 < Ksteps) {
            const __bf16* pa = A + (size_t)(rowBase + (tid >> 1)) * lda +
                               (ks + 1) * 32 + (tid & 1) * 16;
            __builtin_prefetch(pa, 0, 1);
        }
        __syncthreads();

        v16bf a = load_a_frag(&sA[0][0], 32, wave * 16, 0, lane);
#pragma unroll
        for (int nt = 0; nt < 4; ++nt) {
            v16bf b = load_bt_frag(&sWt[0][0], 32, nt * 16, 0, lane);
            acc[nt] = wmma_bf16(a, b, acc[nt]);
        }
    }

    const int hi = lane >> 4, cl = lane & 15;
#pragma unroll
    for (int nt = 0; nt < 4; ++nt) {
        int col = nBase + nt * 16 + cl;
        float bv = bias[col];
#pragma unroll
        for (int q = 0; q < 8; ++q) {
            int row = rowBase + wave * 16 + q + hi * 8;
            float v = acc[nt][q] + bv;
            if (do_relu) v = fmaxf(v, 0.0f);
            out[(size_t)row * ldo + col] = (__bf16)v;
        }
    }
}

// ---------------------------------------------------------------------------
// 5) Final 256 -> 1 layer: dot product per row, wave32 shuffle reduce, f32 out.
// ---------------------------------------------------------------------------
__global__ __launch_bounds__(256) void final_kernel(
    const unsigned short* __restrict__ A_u16, const float* __restrict__ w,
    const float* __restrict__ bias, float* __restrict__ out) {

    const __bf16* A = reinterpret_cast<const __bf16*>(A_u16);
    int lane = threadIdx.x & 31;
    int row  = blockIdx.x * 8 + (threadIdx.x >> 5);
    const __bf16* a = A + (size_t)row * 256;

    float s = 0.0f;
#pragma unroll
    for (int i = 0; i < 8; ++i) {
        int k = lane + 32 * i;
        s += (float)a[k] * w[k];
    }
#pragma unroll
    for (int off = 16; off > 0; off >>= 1) s += __shfl_down(s, off);
    if (lane == 0) out[row] = s + bias[0];
}

// ---------------------------------------------------------------------------
// Host launcher
// ---------------------------------------------------------------------------
extern "C" void kernel_launch(void* const* d_in, const int* in_sizes, int n_in,
                              void* d_out, int out_size, void* d_ws,
                              size_t ws_size, hipStream_t stream) {
    (void)in_sizes; (void)n_in; (void)out_size; (void)ws_size;

    const float* x    = (const float*)d_in[0];
    const float* emb  = (const float*)d_in[1];
    const float* bw0  = (const float*)d_in[2];
    const float* bb0  = (const float*)d_in[3];
    const float* bw1  = (const float*)d_in[4];
    const float* bb1  = (const float*)d_in[5];
    const float* bw2  = (const float*)d_in[6];
    const float* bb2  = (const float*)d_in[7];
    const float* tw0  = (const float*)d_in[8];
    const float* tb0  = (const float*)d_in[9];
    const float* tw1  = (const float*)d_in[10];
    const float* tb1  = (const float*)d_in[11];
    const float* tw2  = (const float*)d_in[12];
    const float* tb2  = (const float*)d_in[13];
    const float* tw3  = (const float*)d_in[14];
    const float* tb3  = (const float*)d_in[15];
    const float* tw4  = (const float*)d_in[16];
    const float* tb4  = (const float*)d_in[17];

    // Workspace carve-up (all bf16 as u16): ~197 MB total.
    char* ws = (char*)d_ws;
    size_t off = 0;
    auto carve = [&](size_t bytes) -> unsigned short* {
        unsigned short* p = (unsigned short*)(ws + off);
        off = (off + bytes + 255) & ~(size_t)255;
        return p;
    };
    unsigned short* combined = carve((size_t)DLRM_B * NCOMB * DLRM_EMBED * 2);
    unsigned short* feat     = carve((size_t)DLRM_B * FEATPAD * 2);
    unsigned short* act1     = carve((size_t)DLRM_B * 1024 * 2);
    unsigned short* act2     = carve((size_t)DLRM_B * 1024 * 2);

    // 1) bottom MLP (also writes combined row 0 + feat[:,0:128] + zero pad)
    bottom_kernel<<<DLRM_B / 32, 64, 0, stream>>>(
        x, bw0, bb0, bw1, bb1, bw2, bb2, combined, feat);

    // 2) embedding gather (bandwidth-bound: 218 MB random reads, ~9.4 us floor)
    gather_kernel<<<DLRM_B * NSPARSE / 8, 256, 0, stream>>>(x, emb, combined);

    // 3) pairwise interaction -> feat[:,128:506]
    interact_kernel<<<DLRM_B / 4, 128, 0, stream>>>(combined, feat);

    // 4) top MLP: 506(512 pad)->1024->1024->512->256, ReLU
    gemm_kernel<<<dim3(DLRM_B / 128, 16), 256, 0, stream>>>(
        feat, FEATPAD, tw0, tb0, 506, 16, 1024, act1, 1024, 1);
    gemm_kernel<<<dim3(DLRM_B / 128, 16), 256, 0, stream>>>(
        act1, 1024, tw1, tb1, 1024, 32, 1024, act2, 1024, 1);
    gemm_kernel<<<dim3(DLRM_B / 128, 8), 256, 0, stream>>>(
        act2, 1024, tw2, tb2, 1024, 32, 512, act1, 512, 1);
    gemm_kernel<<<dim3(DLRM_B / 128, 4), 256, 0, stream>>>(
        act1, 512, tw3, tb3, 512, 16, 256, act2, 256, 1);

    // 5) 256 -> 1, no ReLU, f32 output
    final_kernel<<<DLRM_B / 8, 256, 0, stream>>>(act2, tw4, tb4, (float*)d_out);
}
